// BiDAF_87411174408748
// MI455X (gfx1250) — matrix operand order repaired and checked by reading, not compiled, not run on hardware
//
#include <hip/hip_runtime.h>
#include <hip/hip_bf16.h>

// BiDAF attention flow for MI455X (gfx1250, wave32).
// N=16, CL=4096, QL=64, D=256. Output = [q2c (16*256) | s (16*4096*64)].
#define NB 16
#define CLEN 4096
#define QLEN 64
#define DDIM 256

typedef __attribute__((ext_vector_type(16))) __bf16 v16bf;
typedef __attribute__((ext_vector_type(8)))  float  v8f;

union Frag {
  v16bf v;
  __bf16 e[16];
  uint4  u[2];
};

__device__ __forceinline__ void split_bf16(float x, __bf16& h, __bf16& l) {
  h = (__bf16)x;                 // RNE high part
  l = (__bf16)(x - (float)h);    // residual (exact range in bf16 exponent)
}

// ---------------------------------------------------------------------------
// Kernel A: s = s_c + s_q + (c .* w_cq) @ q^T   (+ all biases), and rowmax(s).
// Grid: (CLEN/128, NB). Block: 256 threads = 8 waves; wave owns 16 rows x 64 cols.
// GEMM via 3x bf16-split WMMA (fp32-grade accuracy, still memory-bound).
// ---------------------------------------------------------------------------
__global__ __launch_bounds__(256)
void bidaf_s_kernel(const float* __restrict__ c,
                    const float* __restrict__ q,
                    const float* __restrict__ w_c,  const float* __restrict__ b_c,
                    const float* __restrict__ w_q,  const float* __restrict__ b_q,
                    const float* __restrict__ w_cq, const float* __restrict__ b_cq,
                    float* __restrict__ s_out,
                    float* __restrict__ rowmax_out)
{
  __shared__ __bf16 qhi[QLEN * DDIM];   // 32 KB
  __shared__ __bf16 qlo[QLEN * DDIM];   // 32 KB
  __shared__ float lds_wc[DDIM];
  __shared__ float lds_wcq[DDIM];
  __shared__ float lds_wq[DDIM];
  __shared__ float lds_sq[QLEN];
  __shared__ float lds_scp[256];

  const int tid    = threadIdx.x;
  const int n      = blockIdx.y;
  const int mBase0 = blockIdx.x * 128;

  // Stage weight vectors (blockDim == DDIM == 256).
  lds_wc[tid]  = w_c[tid];
  lds_wcq[tid] = w_cq[tid];
  lds_wq[tid]  = w_q[tid];
  __syncthreads();

  // Stage q[n] into LDS, pre-split into bf16 hi/lo.
  const float* qn = q + (size_t)n * QLEN * DDIM;
#pragma unroll
  for (int j = 0; j < 16; ++j) {
    int idx4 = tid + j * 256;                       // float4 index (4096 total)
    float4 v = ((const float4*)qn)[idx4];
    int e = idx4 * 4;
    __bf16 h0,l0,h1,l1,h2,l2,h3,l3;
    split_bf16(v.x,h0,l0); split_bf16(v.y,h1,l1);
    split_bf16(v.z,h2,l2); split_bf16(v.w,h3,l3);
    qhi[e+0]=h0; qhi[e+1]=h1; qhi[e+2]=h2; qhi[e+3]=h3;
    qlo[e+0]=l0; qlo[e+1]=l1; qlo[e+2]=l2; qlo[e+3]=l3;
  }

  // s_q[ql] (+ b_q + b_cq folded in); one thread per q row.
  if (tid < QLEN) {
    const float* qr = qn + tid * DDIM;
    float acc = 0.f;
#pragma unroll 4
    for (int k = 0; k < DDIM; k += 4) {
      float4 v = *(const float4*)(qr + k);
      acc += v.x*lds_wq[k] + v.y*lds_wq[k+1] + v.z*lds_wq[k+2] + v.w*lds_wq[k+3];
    }
    lds_sq[tid] = acc + b_q[0] + b_cq[0];
  }
  __syncthreads();

  const int wave = tid >> 5;
  const int lane = tid & 31;
  const int half = lane >> 4;      // half-wave selector (K interleave per ISA layout)
  const int l15  = lane & 15;
  const int mBase = mBase0 + wave * 16;

  // A-fragment source row for this lane (both half-waves cover row l15).
  const float* arow = c + ((size_t)n * CLEN + mBase + l15) * DDIM;
  const float bcv = b_c[0];

  v8f z = {0.f,0.f,0.f,0.f,0.f,0.f,0.f,0.f};
  v8f accs[4] = {z, z, z, z};
  float scp = 0.f;   // partial s_c for row l15 over this lane's K subset

  for (int k0 = 0; k0 < DDIM; k0 += 32) {
    const int ka = k0 + 8 * half;   // A: lo lanes K {0..7,16..23}, hi lanes {8..15,24..31}
    if (k0 + 32 < DDIM) __builtin_prefetch(arow + ka + 32, 0, 3);

    float4 x0 = *(const float4*)(arow + ka);
    float4 x1 = *(const float4*)(arow + ka + 4);
    float4 x2 = *(const float4*)(arow + ka + 16);
    float4 x3 = *(const float4*)(arow + ka + 20);

    // s_c partial (raw c dot w_c) BEFORE scaling by w_cq.
    float4 c0 = *(const float4*)(&lds_wc[ka]);
    float4 c1 = *(const float4*)(&lds_wc[ka + 4]);
    float4 c2 = *(const float4*)(&lds_wc[ka + 16]);
    float4 c3 = *(const float4*)(&lds_wc[ka + 20]);
    scp += x0.x*c0.x + x0.y*c0.y + x0.z*c0.z + x0.w*c0.w
         + x1.x*c1.x + x1.y*c1.y + x1.z*c1.z + x1.w*c1.w
         + x2.x*c2.x + x2.y*c2.y + x2.z*c2.z + x2.w*c2.w
         + x3.x*c3.x + x3.y*c3.y + x3.z*c3.z + x3.w*c3.w;

    // Scale rows by w_cq to form GEMM A operand.
    float4 g0 = *(const float4*)(&lds_wcq[ka]);
    float4 g1 = *(const float4*)(&lds_wcq[ka + 4]);
    float4 g2 = *(const float4*)(&lds_wcq[ka + 16]);
    float4 g3 = *(const float4*)(&lds_wcq[ka + 20]);
    x0.x*=g0.x; x0.y*=g0.y; x0.z*=g0.z; x0.w*=g0.w;
    x1.x*=g1.x; x1.y*=g1.y; x1.z*=g1.z; x1.w*=g1.w;
    x2.x*=g2.x; x2.y*=g2.y; x2.z*=g2.z; x2.w*=g2.w;
    x3.x*=g3.x; x3.y*=g3.y; x3.z*=g3.z; x3.w*=g3.w;

    Frag ah, al;
    split_bf16(x0.x, ah.e[0],  al.e[0]);  split_bf16(x0.y, ah.e[1],  al.e[1]);
    split_bf16(x0.z, ah.e[2],  al.e[2]);  split_bf16(x0.w, ah.e[3],  al.e[3]);
    split_bf16(x1.x, ah.e[4],  al.e[4]);  split_bf16(x1.y, ah.e[5],  al.e[5]);
    split_bf16(x1.z, ah.e[6],  al.e[6]);  split_bf16(x1.w, ah.e[7],  al.e[7]);
    split_bf16(x2.x, ah.e[8],  al.e[8]);  split_bf16(x2.y, ah.e[9],  al.e[9]);
    split_bf16(x2.z, ah.e[10], al.e[10]); split_bf16(x2.w, ah.e[11], al.e[11]);
    split_bf16(x3.x, ah.e[12], al.e[12]); split_bf16(x3.y, ah.e[13], al.e[13]);
    split_bf16(x3.z, ah.e[14], al.e[14]); split_bf16(x3.w, ah.e[15], al.e[15]);

    const int kb = k0 + 8 * half;   // B: lo lanes K {0..7,16..23}, hi lanes {8..15,24..31}
#pragma unroll
    for (int t = 0; t < 4; ++t) {
      const int ql = t * 16 + l15;  // lane = column N of B tile
      const __bf16* ph = &qhi[ql * DDIM + kb];
      const __bf16* pl = &qlo[ql * DDIM + kb];
      Frag bh, bl;
      bh.u[0] = *(const uint4*)(ph);
      bh.u[1] = *(const uint4*)(ph + 16);
      bl.u[0] = *(const uint4*)(pl);
      bl.u[1] = *(const uint4*)(pl + 16);
      // 3-term split: hi*lo + lo*hi + hi*hi, small terms first into the chain.
      accs[t] = __builtin_amdgcn_wmma_f32_16x16x32_bf16(
          false, ah.v, false, bl.v, (short)0, accs[t], false, false);
      accs[t] = __builtin_amdgcn_wmma_f32_16x16x32_bf16(
          false, al.v, false, bh.v, (short)0, accs[t], false, false);
      accs[t] = __builtin_amdgcn_wmma_f32_16x16x32_bf16(
          false, ah.v, false, bh.v, (short)0, accs[t], false, false);
    }
  }

  // Reduce s_c partials: rows' K-halves live in lanes r and r+16.
  lds_scp[tid] = scp;
  __syncthreads();

  float scv[8];
#pragma unroll
  for (int v = 0; v < 8; ++v) {
    int r = v + 8 * half;           // row of accumulator VGPR v in this half-wave
    scv[v] = lds_scp[wave * 32 + r] + lds_scp[wave * 32 + r + 16] + bcv;
  }
  float sqv[4];
#pragma unroll
  for (int t = 0; t < 4; ++t) sqv[t] = lds_sq[t * 16 + l15];

  // Emit s tile + per-row max. C/D layout: VGPR v => row (v + 8*half), col l15.
  float* srow = s_out + ((size_t)n * CLEN + mBase) * QLEN;
  float rmax[8];
#pragma unroll
  for (int v = 0; v < 8; ++v) rmax[v] = -3.402823466e38f;
#pragma unroll
  for (int t = 0; t < 4; ++t) {
#pragma unroll
    for (int v = 0; v < 8; ++v) {
      float val = accs[t][v] + scv[v] + sqv[t];
      rmax[v] = fmaxf(rmax[v], val);
      int row = v + 8 * half;
      srow[(size_t)row * QLEN + t * 16 + l15] = val;
    }
  }
#pragma unroll
  for (int v = 0; v < 8; ++v) {
    float m = rmax[v];
    m = fmaxf(m, __shfl_xor(m, 1, 32));
    m = fmaxf(m, __shfl_xor(m, 2, 32));
    m = fmaxf(m, __shfl_xor(m, 4, 32));
    m = fmaxf(m, __shfl_xor(m, 8, 32));   // stays within each 16-lane half
    if (l15 == 0)
      rowmax_out[(size_t)n * CLEN + mBase + v + 8 * half] = m;
  }
}

// ---------------------------------------------------------------------------
// Kernel B: softmax over CL of rowmax, then q2c[d] = sum_cl b_att[cl]*c[n,cl,d].
// One block per batch; thread == d (DDIM == 256). c re-read hits 192MB L2.
// ---------------------------------------------------------------------------
__global__ __launch_bounds__(256)
void bidaf_softmax_q2c(const float* __restrict__ c,
                       const float* __restrict__ rowmax,
                       float* __restrict__ q2c_out)
{
  __shared__ float w[CLEN];    // 16 KB
  __shared__ float red[256];

  const int tid = threadIdx.x;
  const int n   = blockIdx.x;
  const float* rm = rowmax + (size_t)n * CLEN;

  float lm = -3.402823466e38f;
  for (int i = tid; i < CLEN; i += 256) {
    float v = rm[i];
    w[i] = v;
    lm = fmaxf(lm, v);
  }
  red[tid] = lm;
  __syncthreads();
  for (int s = 128; s > 0; s >>= 1) {
    if (tid < s) red[tid] = fmaxf(red[tid], red[tid + s]);
    __syncthreads();
  }
  const float M = red[0];
  __syncthreads();

  float ls = 0.f;
  for (int i = tid; i < CLEN; i += 256) {
    float e = expf(w[i] - M);
    w[i] = e;
    ls += e;
  }
  red[tid] = ls;
  __syncthreads();
  for (int s = 128; s > 0; s >>= 1) {
    if (tid < s) red[tid] += red[tid + s];
    __syncthreads();
  }
  const float inv = 1.0f / red[0];
  __syncthreads();

  const int d = tid;
  const float* cn = c + (size_t)n * CLEN * DDIM + d;
  float acc = 0.f;
#pragma unroll 8
  for (int cl = 0; cl < CLEN; ++cl)
    acc += w[cl] * cn[(size_t)cl * DDIM];
  q2c_out[n * DDIM + d] = acc * inv;
}

extern "C" void kernel_launch(void* const* d_in, const int* in_sizes, int n_in,
                              void* d_out, int out_size, void* d_ws, size_t ws_size,
                              hipStream_t stream) {
  const float* c    = (const float*)d_in[0];
  const float* q    = (const float*)d_in[1];
  const float* w_c  = (const float*)d_in[2];
  const float* b_c  = (const float*)d_in[3];
  const float* w_q  = (const float*)d_in[4];
  const float* b_q  = (const float*)d_in[5];
  const float* w_cq = (const float*)d_in[6];
  const float* b_cq = (const float*)d_in[7];

  float* out    = (float*)d_out;
  float* q2c    = out;                       // [16*256]
  float* s      = out + (size_t)NB * DDIM;   // [16*4096*64]
  float* rowmax = (float*)d_ws;              // 16*4096 floats (256 KB scratch)

  dim3 g2(CLEN / 128, NB);
  bidaf_s_kernel<<<g2, 256, 0, stream>>>(c, q, w_c, b_c, w_q, b_q, w_cq, b_cq,
                                         s, rowmax);
  bidaf_softmax_q2c<<<NB, 256, 0, stream>>>(c, rowmax, q2c);
}
